// NeuralRadianceField_10479720202678
// MI455X (gfx1250) — matrix operand order repaired and checked by reading, not compiled, not run on hardware
//
#include <hip/hip_runtime.h>
#include <hip/hip_bf16.h>

typedef _Float16 v16h __attribute__((ext_vector_type(16)));
typedef _Float16 v8h  __attribute__((ext_vector_type(8)));
typedef float    v8f  __attribute__((ext_vector_type(8)));

#define STRIDE 336              // halfs per activation row
#define ROWS_PER_BLOCK 64       // 4 waves x 16 rows == one ray (P=64)
#define NPOINTS (2048*64)
#define SIGMA_OFF (NPOINTS*3)

// workspace offsets (in halfs) for f16 fragment-ordered weights
#define OFF0 0u          /* K64 x N256  -> 2*16*512  */
#define OFF1 16384u      /* K256xN256   -> 8*16*512  */
#define OFF2 81920u
#define OFF3 147456u
#define OFF4 212992u
#define OFF5 278528u     /* K320xN256   -> 10*16*512 */
#define OFF6 360448u
#define OFF7 425984u     /* K256xN272   -> 8*17*512  */
#define OFFC 495616u     /* K288xN128   -> 9*8*512   */
#define OFFO 532480u     /* K128xN16    -> 4*1*512   */

__device__ __forceinline__ void wait_lds() {
  asm volatile("s_wait_dscnt 0" ::: "memory");
}

// Convert fp32 weight matrix W[K][N] into f16 WMMA-B fragment order.
// Fragment block (ct,ks) is 512 halfs; lane l holds 16 contiguous halfs:
//   n = ct*16 + (l&15),  k = ks*32 + (l>>4)*8 + (i&7) + (i>>3)*16   (16-bit B layout)
__global__ void convert_weights(const float* __restrict__ W,
                                _Float16* __restrict__ dst,
                                int K, int N, int KS, int CT) {
  int e = blockIdx.x * 256 + threadIdx.x;
  int total = KS * CT * 512;
  if (e >= total) return;
  int i    = e & 15;
  int lane = (e >> 4) & 31;
  int blk  = e >> 9;
  int ks   = blk % KS;
  int ct   = blk / KS;
  int n = ct * 16 + (lane & 15);
  int k = ks * 32 + (lane >> 4) * 8 + (i & 7) + (i >> 3) * 16;
  float v = (k < K && n < N) ? W[k * N + n] : 0.0f;
  dst[e] = (_Float16)v;
}

// One fused layer for a 16-row tile owned by one wave.
// A from LDS act buffer (cols [ASHIFT, ASHIFT+32*KS)), B fragments from
// global, bias-seeded f32 accumulators, optional ReLU, results written back
// to act cols [0, 16*CT).
template<int KS, int CT, int NOUT, bool RELU, int ASHIFT>
__device__ __forceinline__ void run_layer(const _Float16* __restrict__ fragW,
                                          const float* __restrict__ bias,
                                          _Float16* act, int lane) {
  const int row = lane & 15;
  const int grp = lane >> 4;
  v8f acc[CT];
#pragma unroll
  for (int ct = 0; ct < CT; ++ct) {
    int n = ct * 16 + row;
    float bv = (n < NOUT) ? bias[n] : 0.0f;
#pragma unroll
    for (int v = 0; v < 8; ++v) acc[ct][v] = bv;
  }
  for (int ks = 0; ks < KS; ++ks) {
    // A fragment: lane holds row (lane&15); k = ks*32 + grp*8 + {0..7, 16..23}
    const int abase = row * STRIDE + ks * 32 + grp * 8 + ASHIFT;
    v16h a;
    if (ASHIFT == 0) {
      v8h lo = *(const v8h*)(act + abase);
      v8h hi = *(const v8h*)(act + abase + 16);
#pragma unroll
      for (int i = 0; i < 8; ++i) { a[i] = lo[i]; a[i + 8] = hi[i]; }
    } else {  // unaligned (skip/concat shift) — element loads
#pragma unroll
      for (int i = 0; i < 8; ++i) { a[i] = act[abase + i]; a[i + 8] = act[abase + 16 + i]; }
    }
    const _Float16* bp = fragW + (size_t)ks * 512 + (size_t)lane * 16;
#pragma unroll
    for (int ct = 0; ct < CT; ++ct) {
      v16h b = *(const v16h*)(bp + (size_t)ct * (KS * 512));
      acc[ct] = __builtin_amdgcn_wmma_f32_16x16x32_f16(
          false, a, false, b, (short)0, acc[ct], false, false);
    }
  }
  // D layout: lane holds col n = ct*16 + (lane&15), rows v + 8*grp
#pragma unroll
  for (int ct = 0; ct < CT; ++ct) {
    int n = ct * 16 + row;
#pragma unroll
    for (int v = 0; v < 8; ++v) {
      float x = acc[ct][v];
      if (RELU) x = fmaxf(x, 0.0f);
      act[(v + grp * 8) * STRIDE + n] = (_Float16)x;
    }
  }
  wait_lds();
}

__global__ void __launch_bounds__(128)
nerf_fused(const float* __restrict__ pts, const float* __restrict__ dirs,
           const _Float16* __restrict__ ws,
           const float* __restrict__ b0, const float* __restrict__ b1,
           const float* __restrict__ b2, const float* __restrict__ b3,
           const float* __restrict__ b4, const float* __restrict__ b5,
           const float* __restrict__ b6, const float* __restrict__ b7,
           const float* __restrict__ bc, const float* __restrict__ bo,
           float* __restrict__ out) {
  __shared__ _Float16 smem[4][16 * STRIDE];
  const int wave = threadIdx.x >> 5;
  const int lane = threadIdx.x & 31;
  _Float16* act = smem[wave];
  const int r0 = blockIdx.x * ROWS_PER_BLOCK + wave * 16;

  // xyz harmonic embedding -> cols 0..38 (pad to 64) and copy at 256..294
  // (pad to 320) for the skip concat before layer 5.
  if (lane < 16) {
    int g = r0 + lane;
    float p0 = pts[g * 3 + 0], p1 = pts[g * 3 + 1], p2 = pts[g * 3 + 2];
    float p[3] = {p0, p1, p2};
    _Float16* rowp = act + lane * STRIDE;
#pragma unroll
    for (int c = 0; c < 3; ++c) {
      float f = 1.0f;
#pragma unroll
      for (int h = 0; h < 6; ++h) {
        float v = p[c] * f;
        _Float16 s = (_Float16)__sinf(v);
        _Float16 co = (_Float16)__cosf(v);
        rowp[c * 6 + h] = s;        rowp[18 + c * 6 + h] = co;
        rowp[256 + c * 6 + h] = s;  rowp[256 + 18 + c * 6 + h] = co;
        f *= 2.0f;
      }
      rowp[36 + c] = (_Float16)p[c];
      rowp[256 + 36 + c] = (_Float16)p[c];
    }
    for (int j = 39; j < 64; ++j) { rowp[j] = (_Float16)0.f; rowp[256 + j] = (_Float16)0.f; }
  }
  wait_lds();

  run_layer< 2, 16, 256, true,  0>(ws + OFF0, b0, act, lane);   // 39->256
  run_layer< 8, 16, 256, true,  0>(ws + OFF1, b1, act, lane);
  run_layer< 8, 16, 256, true,  0>(ws + OFF2, b2, act, lane);
  run_layer< 8, 16, 256, true,  0>(ws + OFF3, b3, act, lane);
  run_layer< 8, 16, 256, true,  0>(ws + OFF4, b4, act, lane);
  run_layer<10, 16, 256, true,  0>(ws + OFF5, b5, act, lane);   // skip: 295->256
  run_layer< 8, 16, 256, true,  0>(ws + OFF6, b6, act, lane);
  run_layer< 8, 17, 257, false, 0>(ws + OFF7, b7, act, lane);   // 256->257, no ReLU

  // sigma = relu(h[:,0]); direction embedding -> cols 257..283 (pad to 288).
  // Color input is read with a +1 column shift, so Wc rows map 1:1 to k.
  if (lane < 16) {
    int g = r0 + lane;
    out[SIGMA_OFF + g] = fmaxf((float)act[lane * STRIDE + 0], 0.0f);
    int ray = blockIdx.x;  // block == one ray (64 samples)
    float d0 = dirs[ray * 3 + 0], d1 = dirs[ray * 3 + 1], d2 = dirs[ray * 3 + 2];
    float d[3] = {d0, d1, d2};
    _Float16* rowp = act + lane * STRIDE + 257;
#pragma unroll
    for (int c = 0; c < 3; ++c) {
      float f = 1.0f;
#pragma unroll
      for (int h = 0; h < 4; ++h) {
        float v = d[c] * f;
        rowp[c * 4 + h]      = (_Float16)__sinf(v);
        rowp[12 + c * 4 + h] = (_Float16)__cosf(v);
        f *= 2.0f;
      }
      rowp[24 + c] = (_Float16)d[c];
    }
    for (int j = 27; j < 32; ++j) rowp[j] = (_Float16)0.f;
  }
  wait_lds();

  run_layer<9, 8, 128, true, 1>(ws + OFFC, bc, act, lane);      // 283->128 (shifted A)

  // final color layer: K=128, one 16-col tile (3 valid), sigmoid, store
  {
    const int row = lane & 15, grp = lane >> 4;
    v8f acc;
    float bv = (row < 3) ? bo[row] : 0.0f;
#pragma unroll
    for (int v = 0; v < 8; ++v) acc[v] = bv;
#pragma unroll
    for (int ks = 0; ks < 4; ++ks) {
      const int abase = row * STRIDE + ks * 32 + grp * 8;
      v8h lo = *(const v8h*)(act + abase);
      v8h hi = *(const v8h*)(act + abase + 16);
      v16h a;
#pragma unroll
      for (int i = 0; i < 8; ++i) { a[i] = lo[i]; a[i + 8] = hi[i]; }
      v16h b = *(const v16h*)(ws + OFFO + ks * 512 + lane * 16);
      acc = __builtin_amdgcn_wmma_f32_16x16x32_f16(
          false, a, false, b, (short)0, acc, false, false);
    }
    if (row < 3) {
#pragma unroll
      for (int v = 0; v < 8; ++v) {
        int g = r0 + v + grp * 8;
        out[g * 3 + row] = 1.0f / (1.0f + __expf(-acc[v]));
      }
    }
  }
}

extern "C" void kernel_launch(void* const* d_in, const int* in_sizes, int n_in,
                              void* d_out, int out_size, void* d_ws, size_t ws_size,
                              hipStream_t stream) {
  (void)in_sizes; (void)n_in; (void)out_size; (void)ws_size;
  const float* pts  = (const float*)d_in[0];
  const float* dirs = (const float*)d_in[1];
  // W_i at 2+2i, b_i at 3+2i (i=0..7); Wc=18, bc=19, Wo=20, bo=21
  _Float16* ws = (_Float16*)d_ws;
  float* out = (float*)d_out;

  struct { int widx; unsigned off; int K, N, KS, CT; } L[10] = {
    { 2, OFF0,  39, 256,  2, 16},
    { 4, OFF1, 256, 256,  8, 16},
    { 6, OFF2, 256, 256,  8, 16},
    { 8, OFF3, 256, 256,  8, 16},
    {10, OFF4, 256, 256,  8, 16},
    {12, OFF5, 295, 256, 10, 16},
    {14, OFF6, 256, 256,  8, 16},
    {16, OFF7, 256, 257,  8, 17},
    {18, OFFC, 283, 128,  9,  8},
    {20, OFFO, 128,   3,  4,  1},
  };
  for (int i = 0; i < 10; ++i) {
    int total = L[i].KS * L[i].CT * 512;
    int grid = (total + 255) / 256;
    convert_weights<<<grid, 256, 0, stream>>>(
        (const float*)d_in[L[i].widx], ws + L[i].off,
        L[i].K, L[i].N, L[i].KS, L[i].CT);
  }

  nerf_fused<<<NPOINTS / ROWS_PER_BLOCK, 128, 0, stream>>>(
      pts, dirs, ws,
      (const float*)d_in[3],  (const float*)d_in[5],
      (const float*)d_in[7],  (const float*)d_in[9],
      (const float*)d_in[11], (const float*)d_in[13],
      (const float*)d_in[15], (const float*)d_in[17],
      (const float*)d_in[19], (const float*)d_in[21],
      out);
}
